// PiZero_16750372455267
// MI455X (gfx1250) — matrix operand order repaired and checked by reading, not compiled, not run on hardware
//
#include <hip/hip_runtime.h>
#include <hip/hip_bf16.h>
#include <math.h>

typedef __bf16 bf16;
typedef __attribute__((ext_vector_type(16))) __bf16 v16bf;
typedef __attribute__((ext_vector_type(8)))  __bf16 v8bf;
typedef __attribute__((ext_vector_type(8)))  float   v8f;

#define B_     2
#define SV_    512
#define ST_    512
#define NA_    64
#define D_     1024
#define H_     16
#define DH_    64
#define DI_    1024
#define FF_    2730
#define DC_    2048
#define V_     32000
#define DEPTH_ 4
#define DA_    32
#define SEQ_   1024
#define NTOK_  1088
#define EPS_   1.1920929e-07f

#define CEILDIV(a,b) (((a)+(b)-1)/(b))

// =====================================================================
// Tiled GEMM:  C[M,N] = A[M,K](f32,rm) x W[K,N](f32,rm)
// - bf16 conversion in registers -> double-buffered LDS
// - v_wmma_f32_16x16x32_bf16 core, one barrier per K-step
// - vectorized staging (float2 for A since K is even; float4 for W since N%4==0)
// - uniform fast path for interior tiles; guards only on edge tiles
// Block tile 64x128, 8 waves in 4(M) x 2(N); each wave owns 16x64.
// =====================================================================
__global__ __launch_bounds__(256) void gemm_kernel(
    const float* __restrict__ A, const float* __restrict__ W,
    float* __restrict__ C, int M, int N, int K,
    const float* __restrict__ bias, const float* __restrict__ residual)
{
    constexpr int TM = 64, TN = 128, TK = 32;
    __shared__ alignas(16) bf16 As[2][TM][TK + 8];   // row stride 80B (16B-mult)
    __shared__ alignas(16) bf16 Ws[2][TN][TK + 8];   // transposed: Ws[..][n][k]

    const int tid  = threadIdx.x;
    const int wave = tid >> 5, lane = tid & 31;
    const int wm = wave >> 1, wn = wave & 1;
    const int m0 = blockIdx.y * TM, n0 = blockIdx.x * TN;

    const int mrow  = wm * 16 + (lane & 15);
    const int nlane = lane & 15;
    const int kbA   = (lane < 16) ? 0 : 8;    // 16-bit A-fragment K base
    const int kbB   = (lane < 16) ? 0 : 8;    // B-fragment K base

    const int nk = CEILDIV(K, TK);

    float2 aR[4];   // A tile: 64x32 f32 = 1024 float2, 4/thread
    float4 wR[4];   // W tile: 32x128 f32 = 1024 float4, 4/thread

    auto load_tiles = [&](int k0) {
        const bool aFast = (m0 + TM <= M) && (k0 + TK <= K);
        const bool wFast = (n0 + TN <= N) && (k0 + TK <= K);
        if (aFast) {
#pragma unroll
            for (int i = 0; i < 4; ++i) {
                int flat2 = tid + i * 256;
                int r = flat2 >> 4, c2 = flat2 & 15;
                aR[i] = *(const float2*)&A[(size_t)(m0 + r) * K + (k0 + c2 * 2)];
            }
        } else {
#pragma unroll
            for (int i = 0; i < 4; ++i) {
                int flat2 = tid + i * 256;
                int r = flat2 >> 4, c2 = flat2 & 15;
                int gm = m0 + r, gk = k0 + c2 * 2;
                if (gm < M && gk < K) aR[i] = *(const float2*)&A[(size_t)gm * K + gk];
                else { aR[i].x = 0.f; aR[i].y = 0.f; }
            }
        }
        if (wFast) {
#pragma unroll
            for (int i = 0; i < 4; ++i) {
                int flat4 = tid + i * 256;
                int kk = flat4 >> 5, c4 = flat4 & 31;
                wR[i] = *(const float4*)&W[(size_t)(k0 + kk) * N + (n0 + c4 * 4)];
            }
        } else {
#pragma unroll
            for (int i = 0; i < 4; ++i) {
                int flat4 = tid + i * 256;
                int kk = flat4 >> 5, c4 = flat4 & 31;
                int gk = k0 + kk, gn = n0 + c4 * 4;
                if (gk < K && gn < N) wR[i] = *(const float4*)&W[(size_t)gk * N + gn];
                else { wR[i].x = 0.f; wR[i].y = 0.f; wR[i].z = 0.f; wR[i].w = 0.f; }
            }
        }
    };

    auto store_tiles = [&](int pb) {
#pragma unroll
        for (int i = 0; i < 4; ++i) {
            int flat2 = tid + i * 256;
            int r = flat2 >> 4, c2 = flat2 & 15;
            As[pb][r][c2 * 2]     = (bf16)aR[i].x;
            As[pb][r][c2 * 2 + 1] = (bf16)aR[i].y;
        }
#pragma unroll
        for (int i = 0; i < 4; ++i) {
            int flat4 = tid + i * 256;
            int kk = flat4 >> 5, c4 = flat4 & 31;
            Ws[pb][c4 * 4 + 0][kk] = (bf16)wR[i].x;
            Ws[pb][c4 * 4 + 1][kk] = (bf16)wR[i].y;
            Ws[pb][c4 * 4 + 2][kk] = (bf16)wR[i].z;
            Ws[pb][c4 * 4 + 3][kk] = (bf16)wR[i].w;
        }
    };

    v8f acc[4];
#pragma unroll
    for (int c = 0; c < 4; ++c) { v8f z = {0,0,0,0,0,0,0,0}; acc[c] = z; }

    load_tiles(0);
    for (int kt = 0; kt < nk; ++kt) {
        const int pb = kt & 1;
        store_tiles(pb);
        __syncthreads();
        if (kt + 1 < nk) load_tiles((kt + 1) * TK);            // overlap with WMMA below
        if (kt + 2 < nk && tid < 64)                           // L2 prefetch hint, 2 tiles ahead
            __builtin_prefetch(&W[(size_t)((kt + 2) * TK) * N + n0 + (tid << 1)], 0, 1);

        v16bf af;
        {
            v8bf lo = *(const v8bf*)&As[pb][mrow][kbA];
            v8bf hi = *(const v8bf*)&As[pb][mrow][kbA + 16];
#pragma unroll
            for (int i = 0; i < 8; ++i) { af[i] = lo[i]; af[i + 8] = hi[i]; }
        }
#pragma unroll
        for (int c = 0; c < 4; ++c) {
            int ncol = wn * 64 + c * 16 + nlane;
            v16bf bfr;
            v8bf blo = *(const v8bf*)&Ws[pb][ncol][kbB];
            v8bf bhi = *(const v8bf*)&Ws[pb][ncol][kbB + 16];
#pragma unroll
            for (int i = 0; i < 8; ++i) { bfr[i] = blo[i]; bfr[i + 8] = bhi[i]; }
            acc[c] = __builtin_amdgcn_wmma_f32_16x16x32_bf16(
                false, af, false, bfr, (short)0, acc[c], false, false);
        }
        // one barrier per step: next iteration writes buffer pb^1 whose readers
        // (iteration kt-1) are already separated by this iteration's barrier.
    }

    const int hi8 = (lane >> 4) << 3;
#pragma unroll
    for (int c = 0; c < 4; ++c) {
        int col = n0 + wn * 64 + c * 16 + nlane;
#pragma unroll
        for (int r = 0; r < 8; ++r) {
            int row = m0 + wm * 16 + r + hi8;
            if (row < M && col < N) {
                float v = acc[c][r];
                if (bias)     v += bias[col];
                if (residual) v += residual[(size_t)row * N + col];
                C[(size_t)row * N + col] = v;
            }
        }
    }
}

// =====================================================================
// Fused flash attention.  qkv buffers: [rows, 3*DI], col = {q|k|v} x (h*64+d)
// Block = (64-query tile, head, batch), 4 waves; wave owns 16 query rows.
// K tile (32 tok x 64 feat) and V tile (transposed) staged in LDS as bf16
// with coalesced float2 loads; all fragments are aligned ds_load_b128 pairs.
// Online softmax with tanh clamp (50), causal-except-action-columns mask.
// =====================================================================
__device__ __forceinline__ const float* qkv_row(const float* qs, const float* qa,
                                                int b, int t)
{
    return (t < SEQ_) ? (qs + ((size_t)b * SEQ_ + t) * (3 * DI_))
                      : (qa + ((size_t)b * NA_ + (t - SEQ_)) * (3 * DI_));
}

__global__ __launch_bounds__(128) void attn_kernel(
    const float* __restrict__ qkvs, const float* __restrict__ qkva,
    float* __restrict__ o_s, float* __restrict__ o_a)
{
    const int qt = blockIdx.x, h = blockIdx.y, b = blockIdx.z;
    const int tid = threadIdx.x;
    const int w = tid >> 5, lane = tid & 31;
    const int mq  = lane & 15;
    const int hi8 = (lane >> 4) << 3;
    const int kbA = (lane < 16) ? 0 : 8;
    const int kbB = (lane < 16) ? 0 : 8;

    __shared__ alignas(16) bf16  Ks[32][72];      // [token][feature], row 144B
    __shared__ alignas(16) bf16  Vt[64][40];      // [feature][token], row 80B
    __shared__ float Srow[4][16][33];
    __shared__ alignas(16) bf16  Pst[4][16][32];
    __shared__ float scl[4][16];

    // ---- Q fragments (2 feature chunks of 32), pre-scaled by DH^-0.5 ----
    const int tq_lane = qt * 64 + w * 16 + mq;
    const float* qrow = qkv_row(qkvs, qkva, b, tq_lane);
    v16bf qf[2];
#pragma unroll
    for (int ch = 0; ch < 2; ++ch)
#pragma unroll
        for (int g = 0; g < 2; ++g)
#pragma unroll
            for (int i2 = 0; i2 < 4; ++i2) {
                float2 v = *(const float2*)&qrow[h * DH_ + ch * 32 + kbA + g * 16 + i2 * 2];
                qf[ch][g * 8 + i2 * 2]     = (bf16)(v.x * 0.125f);
                qf[ch][g * 8 + i2 * 2 + 1] = (bf16)(v.y * 0.125f);
            }

    v8f O[4];
#pragma unroll
    for (int c = 0; c < 4; ++c) { v8f z = {0,0,0,0,0,0,0,0}; O[c] = z; }
    float mstat = -1e30f, lstat = 0.f;

    for (int j0 = 0; j0 < NTOK_; j0 += 32) {
        // ---- cooperative K/V staging (coalesced float2, bf16 in LDS) ----
#pragma unroll
        for (int i = 0; i < 8; ++i) {
            int flat2 = tid + i * 128;            // 0..1023 over 32 tok x 32 float2
            int tok = flat2 >> 5;
            int f2  = flat2 & 31;
            const float* kr = qkv_row(qkvs, qkva, b, j0 + tok);
            float2 kv2 = *(const float2*)&kr[DI_ + h * DH_ + f2 * 2];
            Ks[tok][f2 * 2]     = (bf16)kv2.x;
            Ks[tok][f2 * 2 + 1] = (bf16)kv2.y;
            float2 vv2 = *(const float2*)&kr[2 * DI_ + h * DH_ + f2 * 2];
            Vt[f2 * 2][tok]     = (bf16)vv2.x;
            Vt[f2 * 2 + 1][tok] = (bf16)vv2.y;
        }
        __syncthreads();

        // ---- S(16x32) = Q(16x64) . K^T ----
        v8f Sh[2];
#pragma unroll
        for (int hc = 0; hc < 2; ++hc) { v8f z = {0,0,0,0,0,0,0,0}; Sh[hc] = z; }
#pragma unroll
        for (int hc = 0; hc < 2; ++hc) {
            int tokl = hc * 16 + mq;              // kv token (local) for this lane's column
#pragma unroll
            for (int ch = 0; ch < 2; ++ch) {
                v16bf kf;
                v8bf lo = *(const v8bf*)&Ks[tokl][ch * 32 + kbB];
                v8bf hi = *(const v8bf*)&Ks[tokl][ch * 32 + kbB + 16];
#pragma unroll
                for (int i = 0; i < 8; ++i) { kf[i] = lo[i]; kf[i + 8] = hi[i]; }
                Sh[hc] = __builtin_amdgcn_wmma_f32_16x16x32_bf16(
                    false, qf[ch], false, kf, (short)0, Sh[hc], false, false);
            }
        }
#pragma unroll
        for (int hc = 0; hc < 2; ++hc)
#pragma unroll
            for (int r = 0; r < 8; ++r)
                Srow[w][r + hi8][hc * 16 + mq] = Sh[hc][r];
        __syncthreads();

        // ---- row owners: clamp, mask, online softmax ----
        if (lane < 16) {
            const int tq = qt * 64 + w * 16 + lane;
            float sv[32];
            float rmax = -1e30f;
#pragma unroll
            for (int c = 0; c < 32; ++c) {
                float x = Srow[w][lane][c];
                x = tanhf(x * 0.02f) * 50.f;                 // softclamp(., 50)
                int tc = j0 + c;
                if (tc > tq && tc < SEQ_) x = -1e30f;        // causal; action cols open
                sv[c] = x;
                rmax = fmaxf(rmax, x);
            }
            float mnew = fmaxf(mstat, rmax);
            float sc = __expf(mstat - mnew);
            float ps = 0.f;
#pragma unroll
            for (int c = 0; c < 32; ++c) {
                float p = __expf(sv[c] - mnew);
                ps += p;
                Pst[w][lane][c] = (bf16)p;
            }
            lstat = lstat * sc + ps;
            mstat = mnew;
            scl[w][lane] = sc;
        }
        __syncthreads();

        // ---- rescale running O, then O(16x64) += P(16x32) . V(32x64) ----
#pragma unroll
        for (int c = 0; c < 4; ++c)
#pragma unroll
            for (int r = 0; r < 8; ++r)
                O[c][r] *= scl[w][r + hi8];

        v16bf pf;
        {
            v8bf lo = *(const v8bf*)&Pst[w][mq][kbA];
            v8bf hi = *(const v8bf*)&Pst[w][mq][kbA + 16];
#pragma unroll
            for (int i = 0; i < 8; ++i) { pf[i] = lo[i]; pf[i + 8] = hi[i]; }
        }
#pragma unroll
        for (int c = 0; c < 4; ++c) {
            v16bf vf;
            v8bf lo = *(const v8bf*)&Vt[c * 16 + mq][kbB];
            v8bf hi = *(const v8bf*)&Vt[c * 16 + mq][kbB + 16];
#pragma unroll
            for (int i = 0; i < 8; ++i) { vf[i] = lo[i]; vf[i + 8] = hi[i]; }
            O[c] = __builtin_amdgcn_wmma_f32_16x16x32_bf16(
                false, pf, false, vf, (short)0, O[c], false, false);
        }
        __syncthreads();   // protect Ks/Vt/Pst for next iteration
    }

    if (lane < 16) scl[w][lane] = 1.f / lstat;
    __syncthreads();

#pragma unroll
    for (int c = 0; c < 4; ++c)
#pragma unroll
        for (int r = 0; r < 8; ++r) {
            int row = r + hi8;
            float v = O[c][r] * scl[w][row];
            int t = qt * 64 + w * 16 + row;
            int feat = h * DH_ + c * 16 + mq;
            if (t < SEQ_) o_s[((size_t)(b * SEQ_ + t)) * DI_ + feat] = v;
            else          o_a[((size_t)(b * NA_ + (t - SEQ_))) * DI_ + feat] = v;
        }
}

// =====================================================================
// Elementwise / reduction kernels
// =====================================================================
__global__ void k_init_out(float* o) { o[0] = 0.f; o[1] = 0.f; }

__global__ __launch_bounds__(256) void k_flow_noised(
    const float* actions, const float* noise, const float* times,
    float* flow, float* na)
{
    int idx = blockIdx.x * 256 + threadIdx.x;
    if (idx >= B_ * NA_ * DA_) return;
    int b = idx / (NA_ * DA_);
    float t = times[b];
    float a = actions[idx], n = noise[idx];
    flow[idx] = a - n;
    na[idx] = n * (1.f - t) + t * a;
}

__global__ __launch_bounds__(256) void k_fourier(
    const float* times, const float* fw, const float* fb, float* rf)
{
    int idx = blockIdx.x * 256 + threadIdx.x;
    if (idx >= B_ * D_) return;
    int b = idx / D_, d = idx % D_;
    rf[idx] = cosf(6.28318530718f * (times[b] * fw[d] + fb[d]));
}

__global__ __launch_bounds__(256) void k_silu(float* x, int n)
{
    int i = blockIdx.x * 256 + threadIdx.x;
    if (i < n) { float v = x[i]; x[i] = v / (1.f + __expf(-v)); }
}

__global__ __launch_bounds__(256) void k_embed(
    const float* images, const int* ids, const float* emb, float* state)
{
    size_t idx = (size_t)blockIdx.x * 256 + threadIdx.x;
    if (idx >= (size_t)B_ * SEQ_ * D_) return;
    int d = idx % D_;
    int p = (idx / D_) % SEQ_;
    int b = idx / ((size_t)D_ * SEQ_);
    float v;
    if (p < SV_) v = images[((size_t)b * SV_ + p) * D_ + d];
    else         v = emb[(size_t)ids[b * ST_ + (p - SV_)] * D_ + d];
    state[idx] = v;
}

// rms with optional soft-clamp; input row = b*inStride + inOff + (row % rowsPerB)
__global__ __launch_bounds__(256) void k_rms(
    const float* __restrict__ x, const float* __restrict__ w, float* __restrict__ y,
    int rowsPerB, int inStride, int inOff, int D, float clampv)
{
    int row = blockIdx.x;
    int b = row / rowsPerB, p = row % rowsPerB;
    const float* xr = x + ((size_t)b * inStride + inOff + p) * D;
    float* yr = y + (size_t)row * D;
    __shared__ float red[256];
    int tid = threadIdx.x;
    float s = 0.f;
    for (int i = tid; i < D; i += 256) {
        float v = xr[i];
        if (clampv > 0.f) v = tanhf(v / clampv) * clampv;
        s += v * v;
    }
    red[tid] = s; __syncthreads();
    for (int off = 128; off > 0; off >>= 1) {
        if (tid < off) red[tid] += red[tid + off];
        __syncthreads();
    }
    float scale = rsqrtf(red[0] / D + EPS_);
    for (int i = tid; i < D; i += 256) {
        float v = xr[i];
        if (clampv > 0.f) v = tanhf(v / clampv) * clampv;
        yr[i] = v * scale * (w ? w[i] : 1.f);
    }
}

// act = rms_na(act) * sigmoid(gpre[b]) + beta[b]   (in place, row-local)
__global__ __launch_bounds__(256) void k_ada(
    float* act, const float* gpre, const float* beta, int rowsPerB, int D)
{
    int row = blockIdx.x;
    int b = row / rowsPerB;
    float* xr = act + (size_t)row * D;
    __shared__ float red[256];
    int tid = threadIdx.x;
    float s = 0.f;
    for (int i = tid; i < D; i += 256) { float v = xr[i]; s += v * v; }
    red[tid] = s; __syncthreads();
    for (int off = 128; off > 0; off >>= 1) {
        if (tid < off) red[tid] += red[tid + off];
        __syncthreads();
    }
    float scale = rsqrtf(red[0] / D + EPS_);
    for (int i = tid; i < D; i += 256) {
        float g = 1.f / (1.f + __expf(-gpre[(size_t)b * D + i]));
        xr[i] = xr[i] * scale * g + beta[(size_t)b * D + i];
    }
}

__global__ __launch_bounds__(256) void k_scale_sig(float* act, const float* spre)
{
    int idx = blockIdx.x * 256 + threadIdx.x;
    if (idx >= B_ * NA_ * D_) return;
    int d = idx % D_;
    int b = idx / (NA_ * D_);
    act[idx] *= 1.f / (1.f + __expf(-spre[b * D_ + d]));
}

__global__ __launch_bounds__(256) void k_swiglu(
    const float* __restrict__ ff, float* __restrict__ hbuf, size_t total)
{
    size_t idx = (size_t)blockIdx.x * 256 + threadIdx.x;
    if (idx >= total) return;
    size_t row = idx / FF_;
    int c = idx % FF_;
    float a = ff[row * (2 * FF_) + c];
    float g = ff[row * (2 * FF_) + FF_ + c];
    float gl = 0.5f * g * (1.f + erff(g * 0.70710678118f));   // exact GELU
    hbuf[idx] = a * gl;
}

__global__ __launch_bounds__(256) void k_lang_loss(
    const float* __restrict__ logits, const int* __restrict__ ids, float* out)
{
    int blk = blockIdx.x;                 // 0 .. B*(ST-1)-1
    int b = blk / (ST_ - 1), i = blk % (ST_ - 1);
    const float* lr = logits + ((size_t)(b * ST_ + i)) * V_;
    __shared__ float red[256];
    int tid = threadIdx.x;
    float mx = -1e30f;
    for (int j = tid; j < V_; j += 256) mx = fmaxf(mx, lr[j]);
    red[tid] = mx; __syncthreads();
    for (int off = 128; off > 0; off >>= 1) {
        if (tid < off) red[tid] = fmaxf(red[tid], red[tid + off]);
        __syncthreads();
    }
    mx = red[0]; __syncthreads();
    float s = 0.f;
    for (int j = tid; j < V_; j += 256) s += __expf(lr[j] - mx);
    red[tid] = s; __syncthreads();
    for (int off = 128; off > 0; off >>= 1) {
        if (tid < off) red[tid] += red[tid + off];
        __syncthreads();
    }
    if (tid == 0) {
        int label = ids[b * ST_ + i + 1];
        float lp = lr[label] - mx - logf(red[0]);
        atomicAdd(out, -lp / (float)(B_ * (ST_ - 1)));
    }
}

__global__ __launch_bounds__(256) void k_flow_loss(
    const float* flow, const float* pred, float* out)
{
    __shared__ float red[256];
    int tid = threadIdx.x;
    float s = 0.f;
    for (int i = tid; i < B_ * NA_ * DA_; i += 256) {
        float d = flow[i] - pred[i];
        s += d * d;
    }
    red[tid] = s; __syncthreads();
    for (int off = 128; off > 0; off >>= 1) {
        if (tid < off) red[tid] += red[tid + off];
        __syncthreads();
    }
    if (tid == 0) out[1] = red[0] / (float)(B_ * NA_ * DA_);
}

// =====================================================================
// Host orchestration
// =====================================================================
static inline void gemm(hipStream_t st, const float* A, const float* W, float* C,
                        int M, int N, int K, const float* bias, const float* res)
{
    dim3 g(CEILDIV(N, 128), CEILDIV(M, 64));
    gemm_kernel<<<g, 256, 0, st>>>(A, W, C, M, N, K, bias, res);
}

extern "C" void kernel_launch(void* const* d_in, const int* in_sizes, int n_in,
                              void* d_out, int out_size, void* d_ws, size_t ws_size,
                              hipStream_t stream)
{
    (void)in_sizes; (void)n_in; (void)out_size; (void)ws_size;

    // setup_inputs() dict order, params/layers flattened in insertion order
    const float* images    = (const float*)d_in[0];
    const int*   token_ids = (const int*)  d_in[1];
    const float* actions   = (const float*)d_in[2];
    const float* times     = (const float*)d_in[3];
    const float* noise     = (const float*)d_in[4];
    const float* token_emb = (const float*)d_in[5];
    const float* act_w     = (const float*)d_in[6];
    const float* act_b     = (const float*)d_in[7];
    const float* four_w    = (const float*)d_in[8];
    const float* four_b    = (const float*)d_in[9];
    const float* time_w    = (const float*)d_in[10];
    const float* time_b    = (const float*)d_in[11];
    const float* final_n   = (const float*)d_in[12];
    const float* final_an  = (const float*)d_in[13];
    const float* logits_w  = (const float*)d_in[14];
    const float* flow_w    = (const float*)d_in[15];
    const float* L_rn_s    = (const float*)d_in[16];
    const float* L_rn_a    = (const float*)d_in[17];
    const float* L_qkv_s   = (const float*)d_in[18];
    const float* L_out_s   = (const float*)d_in[19];
    const float* L_qkv_a   = (const float*)d_in[20];
    const float* L_out_a   = (const float*)d_in[21];
    const float* L_ffs_rn  = (const float*)d_in[22];
    const float* L_ffs_in  = (const float*)d_in[23];
    const float* L_ffs_out = (const float*)d_in[24];
    const float* L_ffa_rn  = (const float*)d_in[25];
    const float* L_ffa_in  = (const float*)d_in[26];
    const float* L_ffa_out = (const float*)d_in[27];
    const float* L_arn_gw  = (const float*)d_in[28];
    const float* L_arn_gb  = (const float*)d_in[29];
    const float* L_arn_bw  = (const float*)d_in[30];
    const float* L_als_w   = (const float*)d_in[31];
    const float* L_als_b   = (const float*)d_in[32];
    const float* L_frn_gw  = (const float*)d_in[33];
    const float* L_frn_gb  = (const float*)d_in[34];
    const float* L_frn_bw  = (const float*)d_in[35];

    float* out = (float*)d_out;
    float* ws  = (float*)d_ws;
    size_t off = 0;
    auto alloc = [&](size_t n) { float* p = ws + off; off += n; return p; };

    float* flow   = alloc((size_t)B_ * NA_ * DA_);
    float* na     = alloc((size_t)B_ * NA_ * DA_);
    float* rf     = alloc((size_t)B_ * D_);
    float* tc     = alloc((size_t)B_ * DC_);
    float* act    = alloc((size_t)B_ * NA_ * D_);
    float* state  = alloc((size_t)B_ * SEQ_ * D_);
    float* gpre   = alloc((size_t)B_ * D_);
    float* beta   = alloc((size_t)B_ * D_);
    float* s_norm = alloc((size_t)B_ * SEQ_ * D_);
    float* a_norm = alloc((size_t)B_ * NA_ * D_);
    float* qkvs   = alloc((size_t)B_ * SEQ_ * 3 * DI_);
    float* qkva   = alloc((size_t)B_ * NA_ * 3 * DI_);
    float* o_s    = alloc((size_t)B_ * SEQ_ * DI_);
    float* o_a    = alloc((size_t)B_ * NA_ * DI_);
    float* ff     = alloc((size_t)B_ * SEQ_ * 2 * FF_);
    float* hbuf   = alloc((size_t)B_ * SEQ_ * FF_);
    float* toksn  = alloc((size_t)B_ * ST_ * D_);
    float* pred   = alloc((size_t)B_ * NA_ * DA_);
    float* logits = alloc((size_t)B_ * ST_ * V_);

    hipStream_t st = stream;
    k_init_out<<<1, 1, 0, st>>>(out);

    // preliminaries
    k_flow_noised<<<CEILDIV(B_*NA_*DA_, 256), 256, 0, st>>>(actions, noise, times, flow, na);
    k_fourier<<<CEILDIV(B_*D_, 256), 256, 0, st>>>(times, four_w, four_b, rf);
    gemm(st, rf, time_w, tc, B_, DC_, D_, time_b, nullptr);
    k_silu<<<CEILDIV(B_*DC_, 256), 256, 0, st>>>(tc, B_ * DC_);
    gemm(st, na, act_w, act, B_ * NA_, D_, DA_, act_b, nullptr);
    k_embed<<<CEILDIV(B_*SEQ_*D_, 256), 256, 0, st>>>(images, token_ids, token_emb, state);

    const int Ms = B_ * SEQ_;   // 2048
    const int Ma = B_ * NA_;    // 128

    for (int l = 0; l < DEPTH_; ++l) {
        const float* rn_s   = L_rn_s   + (size_t)l * D_;
        const float* rn_a   = L_rn_a   + (size_t)l * D_;
        const float* qkv_sw = L_qkv_s  + (size_t)l * D_ * 3 * DI_;
        const float* out_sw = L_out_s  + (size_t)l * DI_ * D_;
        const float* qkv_aw = L_qkv_a  + (size_t)l * D_ * 3 * DI_;
        const float* out_aw = L_out_a  + (size_t)l * DI_ * D_;
        const float* ffs_rn = L_ffs_rn + (size_t)l * D_;
        const float* ffs_in = L_ffs_in + (size_t)l * D_ * 2 * FF_;
        const float* ffs_ou = L_ffs_out+ (size_t)l * FF_ * D_;
        const float* ffa_rn = L_ffa_rn + (size_t)l * D_;
        const float* ffa_in = L_ffa_in + (size_t)l * D_ * 2 * FF_;
        const float* ffa_ou = L_ffa_out+ (size_t)l * FF_ * D_;
        const float* arn_gw = L_arn_gw + (size_t)l * DC_ * D_;
        const float* arn_gb = L_arn_gb + (size_t)l * D_;
        const float* arn_bw = L_arn_bw + (size_t)l * DC_ * D_;
        const float* als_w  = L_als_w  + (size_t)l * DC_ * D_;
        const float* als_b  = L_als_b  + (size_t)l * D_;
        const float* frn_gw = L_frn_gw + (size_t)l * DC_ * D_;
        const float* frn_gb = L_frn_gb + (size_t)l * D_;
        const float* frn_bw = L_frn_bw + (size_t)l * DC_ * D_;

        // act = ada_rmsnorm(act, time_cond, arn)
        gemm(st, tc, arn_gw, gpre, B_, D_, DC_, arn_gb, nullptr);
        gemm(st, tc, arn_bw, beta, B_, D_, DC_, nullptr, nullptr);
        k_ada<<<Ma, 256, 0, st>>>(act, gpre, beta, NA_, D_);

        // attention
        k_rms<<<Ms, 256, 0, st>>>(state, rn_s, s_norm, Ms, 0, 0, D_, 0.f);
        k_rms<<<Ma, 256, 0, st>>>(act, rn_a, a_norm, Ma, 0, 0, D_, 0.f);
        gemm(st, s_norm, qkv_sw, qkvs, Ms, 3 * DI_, D_, nullptr, nullptr);
        gemm(st, a_norm, qkv_aw, qkva, Ma, 3 * DI_, D_, nullptr, nullptr);
        attn_kernel<<<dim3(NTOK_ / 64, H_, B_), 128, 0, st>>>(qkvs, qkva, o_s, o_a);

        // act = ada_layerscale(act) ; residual adds fused in out-proj GEMMs
        gemm(st, tc, als_w, gpre, B_, D_, DC_, als_b, nullptr);
        k_scale_sig<<<CEILDIV(B_*NA_*D_, 256), 256, 0, st>>>(act, gpre);
        gemm(st, o_s, out_sw, state, Ms, D_, DI_, nullptr, state);
        gemm(st, o_a, out_aw, act,   Ma, D_, DI_, nullptr, act);

        // state = swiglu(state) + state
        k_rms<<<Ms, 256, 0, st>>>(state, ffs_rn, s_norm, Ms, 0, 0, D_, 0.f);
        gemm(st, s_norm, ffs_in, ff, Ms, 2 * FF_, D_, nullptr, nullptr);
        k_swiglu<<<CEILDIV((size_t)Ms*FF_, (size_t)256), 256, 0, st>>>(ff, hbuf, (size_t)Ms * FF_);
        gemm(st, hbuf, ffs_ou, state, Ms, D_, FF_, nullptr, state);

        // act = ada_rmsnorm(act, frn)
        gemm(st, tc, frn_gw, gpre, B_, D_, DC_, frn_gb, nullptr);
        gemm(st, tc, frn_bw, beta, B_, D_, DC_, nullptr, nullptr);
        k_ada<<<Ma, 256, 0, st>>>(act, gpre, beta, NA_, D_);

        // act = swiglu(act) + act
        k_rms<<<Ma, 256, 0, st>>>(act, ffa_rn, a_norm, Ma, 0, 0, D_, 0.f);
        gemm(st, a_norm, ffa_in, ff, Ma, 2 * FF_, D_, nullptr, nullptr);
        k_swiglu<<<CEILDIV((size_t)Ma*FF_, (size_t)256), 256, 0, st>>>(ff, hbuf, (size_t)Ma * FF_);
        gemm(st, hbuf, ffa_ou, act, Ma, D_, FF_, nullptr, act);

        // act = ada_rmsnorm(act, frn)  (same conditioners, gpre/beta still valid)
        k_ada<<<Ma, 256, 0, st>>>(act, gpre, beta, NA_, D_);
    }

    // final heads
    // toks = rms(softclamp(state[:, SV:], 30), final_norm)  -> contiguous (B*ST, D)
    k_rms<<<B_ * ST_, 256, 0, st>>>(state, final_n, toksn, ST_, SEQ_, SV_, D_, 30.f);
    gemm(st, toksn, logits_w, logits, B_ * ST_, V_, D_, nullptr, nullptr);
    k_lang_loss<<<B_ * (ST_ - 1), 256, 0, st>>>(logits, token_ids, out);

    // act head
    k_rms<<<Ma, 256, 0, st>>>(act, final_an, a_norm, Ma, 0, 0, D_, 30.f);
    gemm(st, a_norm, flow_w, pred, Ma, DA_, D_, nullptr, nullptr);
    k_flow_loss<<<1, 256, 0, st>>>(flow, pred, out);
}
